// RandomProjectionQuantizer_46334107189359
// MI455X (gfx1250) — matrix operand order repaired and checked by reading, not compiled, not run on hardware
//
#include <hip/hip_runtime.h>
#include <hip/hip_bf16.h>

// ---------------------------------------------------------------------------
// RandomProjectionQuantizer for MI455X (gfx1250, wave32, WMMA + TDM).
//
// codes[c,b,t] = argmax_k  <mel_t[b,t,:] @ proj[c], cb_norm[c,k]>
//
// - L2-normalizing the projected vector is argmax-invariant (positive per-row
//   scale) -> skipped. Codebook norms folded into a one-time prep pass that
//   also pre-swizzles codebooks into the WMMA bf16 B-fragment layout.
// - Both GEMMs run on v_wmma_f32_16x16x32_bf16.
// - The strided mel tile ([80 bins] x [16 t], row stride 4000 floats) is
//   staged into LDS by the Tensor Data Mover: one tensor_load_to_lds per
//   wave with a 2D descriptor, tracked on TENSORcnt.
// ---------------------------------------------------------------------------

typedef __attribute__((ext_vector_type(16))) __bf16        v16bf;
typedef __attribute__((ext_vector_type(8)))  float         v8f;
typedef __attribute__((ext_vector_type(4)))  unsigned int  u32x4;
typedef __attribute__((ext_vector_type(8)))  unsigned int  u32x8;

union FragU {
    unsigned int u[8];
    v16bf        v;
};

__device__ __forceinline__ unsigned short bf16_bits(float f) {
    unsigned int u = __float_as_uint(f);
    return (unsigned short)((u + 0x7FFFu + ((u >> 16) & 1u)) >> 16);  // RNE
}

__device__ __forceinline__ unsigned int pack_bf16(float lo, float hi) {
    return ((unsigned int)bf16_bits(hi) << 16) | (unsigned int)bf16_bits(lo);
}

// Problem constants (from setup_inputs)
#define NB   32      // batch
#define NMEL 80      // mel bins (GEMM1 K, padded to 96)
#define NT   4000    // time steps
#define NC   2       // codebooks
#define NK   1024    // codebook size
#define ND   64      // d (GEMM1 N / GEMM2 K)

#define TILES_PER_T   (NT / 16)            // 250 row-tiles per batch entry
#define TILES_PER_CB  (NB * TILES_PER_T)   // 8000 row-tiles per codebook
#define WAVES_PER_BLK 8

// B-fragment dword layout (our convention, consistent across prep + main):
//   index = ((tileLinear * 32) + lane) * 8 + r     (8 dwords contiguous/lane)
//   lane: n = tile_n*16 + (lane & 15), khalf = lane >> 4
//   dword r holds bf16 pair for K = kchunk*32 + khalf*16 + 2r (lo), +1 (hi)

// ---- prep: normalize codebooks -> bf16 B fragments --------------------------
__global__ __launch_bounds__(256) void rpq_prep_cb(const float* __restrict__ cb,
                                                   unsigned int* __restrict__ cbB) {
    int tid = blockIdx.x * blockDim.x + threadIdx.x;           // 65536 total
    if (tid >= NC * 64 * 2 * 32 * 8) return;
    int r    = tid & 7;
    int lane = (tid >> 3) & 31;
    int kc   = (tid >> 8) & 1;
    int kt   = (tid >> 9) & 63;
    int c    = tid >> 15;
    int n  = kt * 16 + (lane & 15);                            // codebook entry
    int kd = kc * 32 + (lane >> 4) * 16 + 2 * r;               // d index 0..62
    const float* rowp = cb + ((size_t)c * NK + n) * ND;
    float s = 0.f;
#pragma unroll
    for (int d = 0; d < ND; ++d) s += rowp[d] * rowp[d];
    float inv = 1.0f / fmaxf(sqrtf(s), 1e-12f);
    cbB[tid] = pack_bf16(rowp[kd] * inv, rowp[kd + 1] * inv);
}

// ---- prep: proj -> bf16 B fragments (K padded 80 -> 96) ---------------------
__global__ __launch_bounds__(256) void rpq_prep_proj(const float* __restrict__ proj,
                                                     unsigned int* __restrict__ projB) {
    int tid = blockIdx.x * blockDim.x + threadIdx.x;           // 6144 total
    if (tid >= NC * 4 * 3 * 32 * 8) return;
    int r    = tid & 7;
    int lane = (tid >> 3) & 31;
    int rem  = tid >> 8;
    int kc   = rem % 3;
    int nt   = (rem / 3) & 3;
    int c    = tid / 3072;
    int n  = nt * 16 + (lane & 15);                            // d column
    int kd = kc * 32 + (lane >> 4) * 16 + 2 * r;               // mel bin 0..94
    float lo = (kd     < NMEL) ? proj[((size_t)c * NMEL + kd)     * ND + n] : 0.f;
    float hi = (kd + 1 < NMEL) ? proj[((size_t)c * NMEL + kd + 1) * ND + n] : 0.f;
    projB[tid] = pack_bf16(lo, hi);
}

// ---- main: TDM stage + project + match + argmax -----------------------------
__global__ __launch_bounds__(256) void rpq_main(const float* __restrict__ mel,
                                                const unsigned int* __restrict__ projB,
                                                const unsigned int* __restrict__ cbB,
                                                int* __restrict__ codes) {
    // mel tile per wave, TDM-written: [n=80][t=16] f32, rows contiguous
    __shared__ float          ldsMel[WAVES_PER_BLK][NMEL][16];
    __shared__ unsigned short ldsP[WAVES_PER_BLK][16 * ND];    // p tile bf16 [m][d]

    const int tid  = threadIdx.x;
    const int w    = tid >> 5;
    const int lane = tid & 31;
    const int row  = lane & 15;   // A-matrix row (t) / D-matrix N index
    const int sel  = lane >> 4;   // lane half

    const int gtile = blockIdx.x * WAVES_PER_BLK + w;          // 0..15999
    const int c  = gtile / TILES_PER_CB;
    const int rt = gtile % TILES_PER_CB;
    const int b  = rt / TILES_PER_T;
    const int t0 = (rt % TILES_PER_T) * 16;

    // ---- TDM: strided 2D gather of the mel tile into LDS -------------------
    {
        const float* gptr = mel + (size_t)b * (NMEL * NT) + t0;
        unsigned long long ga = (unsigned long long)(uintptr_t)gptr;
        unsigned ga_lo  = __builtin_amdgcn_readfirstlane((unsigned)ga);
        unsigned ga_hi  = __builtin_amdgcn_readfirstlane((unsigned)(ga >> 32));
        unsigned ldsOff = __builtin_amdgcn_readfirstlane(
            (unsigned)(uintptr_t)&ldsMel[w][0][0]);   // flat[31:0] == LDS offset

        u32x4 g0;
        g0[0] = 1u;                                    // count=1, user descriptor
        g0[1] = ldsOff;                                // lds_addr (bytes)
        g0[2] = ga_lo;                                 // global_addr[31:0]
        g0[3] = (ga_hi & 0x01FFFFFFu) | 0x80000000u;   // addr[56:32] | type=2

        u32x8 g1;
        g1[0] = 0x00020000u;          // wg_mask=0, data_size=2 (4B), no flags
        g1[1] = 16u << 16;            // tensor_dim0[15:0]=16 (t extent)
        g1[2] = (unsigned)NMEL << 16; // tensor_dim0 hi=0 | tensor_dim1 lo=80
        g1[3] = 16u << 16;            // tensor_dim1 hi=0 | tile_dim0=16
        g1[4] = (unsigned)NMEL;       // tile_dim1=80, tile_dim2=0
        g1[5] = (unsigned)NT;         // tensor_dim0_stride lo = 4000 elements
        g1[6] = 0u;                   // stride hi=0 | dim1_stride lo=0
        g1[7] = 0u;                   // dim1_stride hi=0

        asm volatile("tensor_load_to_lds %0, %1"
                     :
                     : "s"(g0), "s"(g1)
                     : "memory");
        __builtin_amdgcn_s_wait_tensorcnt(0);          // TDM done -> LDS valid
    }

    // ---- build bf16 A fragments for GEMM1 (16x32 per chunk, K pad to 96) ---
    const float* melT = (const float*)&ldsMel[w][0][0];   // [k=80][t=16]
    FragU amel[3];
#pragma unroll
    for (int kc = 0; kc < 3; ++kc) {
#pragma unroll
        for (int r = 0; r < 8; ++r) {
            int k = kc * 32 + sel * 16 + 2 * r;
            float lo = (k     < NMEL) ? melT[k * 16 + row]       : 0.f;
            float hi = (k + 1 < NMEL) ? melT[(k + 1) * 16 + row] : 0.f;
            amel[kc].u[r] = pack_bf16(lo, hi);
        }
    }

    // ---- GEMM1: P[16x64] = A[16x96] x proj[96x64]  (12 WMMAs) --------------
#pragma unroll
    for (int nt = 0; nt < 4; ++nt) {
        v8f acc = {0.f, 0.f, 0.f, 0.f, 0.f, 0.f, 0.f, 0.f};
#pragma unroll
        for (int kc = 0; kc < 3; ++kc) {
            const uint4* bp =
                (const uint4*)(projB + ((((size_t)c * 4 + nt) * 3 + kc) * 32 + lane) * 8);
            uint4 b0 = bp[0], b1 = bp[1];
            FragU bf;
            bf.u[0] = b0.x; bf.u[1] = b0.y; bf.u[2] = b0.z; bf.u[3] = b0.w;
            bf.u[4] = b1.x; bf.u[5] = b1.y; bf.u[6] = b1.z; bf.u[7] = b1.w;
            acc = __builtin_amdgcn_wmma_f32_16x16x32_bf16(
                false, amel[kc].v, false, bf.v, (short)0, acc, false, false);
        }
        // D layout: vgpr i -> M = sel*8 + i, N = lane&15. Store bf16 [m][d].
        const int col = nt * 16 + row;
#pragma unroll
        for (int i = 0; i < 8; ++i)
            ldsP[w][(sel * 8 + i) * ND + col] = bf16_bits(acc[i]);
    }
    __syncthreads();   // cross-lane visibility of ldsP (uniform across block)

    // ---- rebuild P as A fragments for GEMM2 (K = 64, two chunks) -----------
    FragU pA[2];
    {
        const unsigned int* pRow = (const unsigned int*)&ldsP[w][0];
#pragma unroll
        for (int kc = 0; kc < 2; ++kc)
#pragma unroll
            for (int r = 0; r < 8; ++r) {
                int k = kc * 32 + sel * 16 + 2 * r;
                pA[kc].u[r] = pRow[(row * ND + k) >> 1];
            }
    }

    // ---- GEMM2 + running argmax over 64 codebook tiles (128 WMMAs) ---------
    float best[8];
    int   bidx[8];
#pragma unroll
    for (int i = 0; i < 8; ++i) { best[i] = -3.402823466e38f; bidx[i] = 0; }

    const unsigned int* cbBase = cbB + (size_t)c * (64 * 2 * 32 * 8);
    for (int kt = 0; kt < 64; ++kt) {
        if (kt + 1 < 64)   // warm next B tile: lowers to global_prefetch_b8
            __builtin_prefetch(cbBase + (((size_t)(kt + 1) * 2) * 32 + lane) * 8, 0, 0);
        v8f acc = {0.f, 0.f, 0.f, 0.f, 0.f, 0.f, 0.f, 0.f};
#pragma unroll
        for (int kc = 0; kc < 2; ++kc) {
            const uint4* bp =
                (const uint4*)(cbBase + (((size_t)kt * 2 + kc) * 32 + lane) * 8);
            uint4 b0 = bp[0], b1 = bp[1];
            FragU bf;
            bf.u[0] = b0.x; bf.u[1] = b0.y; bf.u[2] = b0.z; bf.u[3] = b0.w;
            bf.u[4] = b1.x; bf.u[5] = b1.y; bf.u[6] = b1.z; bf.u[7] = b1.w;
            acc = __builtin_amdgcn_wmma_f32_16x16x32_bf16(
                false, pA[kc].v, false, bf.v, (short)0, acc, false, false);
        }
        const int kbase = kt * 16 + row;   // this lane's codebook index
#pragma unroll
        for (int i = 0; i < 8; ++i) {
            float v = acc[i];
            if (v > best[i]) { best[i] = v; bidx[i] = kbase; }  // strict: first max wins
        }
    }

    // ---- argmax reduction across the 16 lanes holding one row --------------
#pragma unroll
    for (int off = 8; off >= 1; off >>= 1) {
#pragma unroll
        for (int i = 0; i < 8; ++i) {
            float ov = __shfl_xor(best[i], off, 16);
            int   oi = __shfl_xor(bidx[i], off, 16);
            if (ov > best[i] || (ov == best[i] && oi < bidx[i])) {
                best[i] = ov;
                bidx[i] = oi;
            }
        }
    }

    // lane 0 writes rows 0..7, lane 16 writes rows 8..15
    if (row == 0) {
        int* outp = codes + (size_t)c * (NB * NT) + (size_t)b * NT + t0 + sel * 8;
#pragma unroll
        for (int i = 0; i < 8; ++i) outp[i] = bidx[i];
    }
}

// ---------------------------------------------------------------------------
extern "C" void kernel_launch(void* const* d_in, const int* in_sizes, int n_in,
                              void* d_out, int out_size, void* d_ws, size_t ws_size,
                              hipStream_t stream) {
    const float* mel  = (const float*)d_in[0];   // [32, 80, 4000] f32
    const float* proj = (const float*)d_in[1];   // [2, 80, 64]    f32
    const float* cb   = (const float*)d_in[2];   // [2, 1024, 64]  f32

    // workspace: 65536 dwords cbB (256 KB) + 6144 dwords projB (24 KB)
    unsigned int* cbB   = (unsigned int*)d_ws;
    unsigned int* projB = cbB + (NC * 64 * 2 * 32 * 8);
    int* codes = (int*)d_out;                    // [2, 32, 4000] int32

    rpq_prep_cb  <<<(NC * 64 * 2 * 32 * 8) / 256, 256, 0, stream>>>(cb, cbB);
    rpq_prep_proj<<<(NC * 4 * 3 * 32 * 8 + 255) / 256, 256, 0, stream>>>(proj, projB);

    const int nTiles = NC * TILES_PER_CB;                 // 16000 wave-tiles
    rpq_main<<<nTiles / WAVES_PER_BLK, 256, 0, stream>>>(mel, projB, cbB, codes);
}
